// NearestNeighborTokenizer_5111011082542
// MI455X (gfx1250) — matrix operand order repaired and checked by reading, not compiled, not run on hardware
//
#include <hip/hip_runtime.h>

#define DIM        64
#define NUM_CODES  50000
#define NSPLIT     8
#define NTOK       8192
#define BM         64
#define BN         64
#define THRESH     100.0f

typedef __attribute__((ext_vector_type(16))) __bf16 v16bf;
typedef __attribute__((ext_vector_type(8)))  __bf16 v8bf;
typedef __attribute__((ext_vector_type(8)))  float  v8f;

typedef unsigned int u32x4 __attribute__((ext_vector_type(4)));
typedef int          i32x8 __attribute__((ext_vector_type(8)));
typedef int          i32x4 __attribute__((ext_vector_type(4)));

union AFrag { v16bf v; v8bf h[2]; };

__device__ __forceinline__ unsigned short bf16_rne(float f) {
    unsigned int u = __float_as_uint(f);
    u += 0x7FFFu + ((u >> 16) & 1u);          // round-to-nearest-even into bf16
    return (unsigned short)(u >> 16);
}

// ---------------------------------------------------------------------------
// Kernel 0: one-time codes -> bf16 hi/lo split (packed) + ||c||^2
// c2 is padded with 64 +INF sentinels so the main loop can load unconditionally.
// ---------------------------------------------------------------------------
__global__ __launch_bounds__(256)
void code_precompute_kernel(const float* __restrict__ codes,
                            unsigned* __restrict__ chi, unsigned* __restrict__ clo,
                            float* __restrict__ c2) {
    const int row  = blockIdx.x * 128 + (threadIdx.x >> 1);
    const int half = threadIdx.x & 1;                 // 32 elements each
    if (row >= NUM_CODES + BN) return;
    if (row >= NUM_CODES) {                           // sentinel tail (TDM zero-fills data)
        if (half == 0) c2[row] = 3.0e38f;
        return;
    }
    const float* src = codes + (size_t)row * DIM + half * 32;
    unsigned* dh = chi + (size_t)row * (DIM / 2) + half * 16;
    unsigned* dl = clo + (size_t)row * (DIM / 2) + half * 16;
    float s2 = 0.0f;
    #pragma unroll
    for (int j = 0; j < 16; ++j) {
        float f0 = src[2 * j], f1 = src[2 * j + 1];
        s2 = __builtin_fmaf(f0, f0, __builtin_fmaf(f1, f1, s2));
        unsigned short h0 = bf16_rne(f0), h1 = bf16_rne(f1);
        float hf0 = __uint_as_float((unsigned)h0 << 16);
        float hf1 = __uint_as_float((unsigned)h1 << 16);
        unsigned short l0 = bf16_rne(f0 - hf0), l1 = bf16_rne(f1 - hf1);
        dh[j] = (unsigned)h0 | ((unsigned)h1 << 16);
        dl[j] = (unsigned)l0 | ((unsigned)l1 << 16);
    }
    float s2o = __shfl_xor(s2, 1, 32);                // partner thread = other half-row
    if (half == 0) c2[row] = s2 + s2o;
}

// ---------------------------------------------------------------------------
// TDM: 2D tile load (64 rows x 128B) from a [NUM_CODES][128B] bf16 array -> LDS
// ---------------------------------------------------------------------------
__device__ __forceinline__ void tdm_load_tile(unsigned long long gaddr, unsigned lds_off) {
    u32x4 g0;
    g0[0] = 1u;                                            // count=1, flags=0
    g0[1] = lds_off;                                       // LDS byte address
    g0[2] = (unsigned)(gaddr & 0xFFFFFFFFu);               // global_addr[31:0]
    g0[3] = (unsigned)((gaddr >> 32) & 0x1FFFFFFu) | (2u << 30);  // addr[56:32] | type=2
    i32x8 g1;
    g1[0] = 0;                                             // mask=0, data_size=1B, no flags
    g1[1] = (int)(128u << 16);                             // tensor_dim0 = 128 bytes
    g1[2] = (int)(((unsigned)NUM_CODES & 0xFFFFu) << 16);  // tensor_dim1.lo16 (dim0.hi16=0)
    g1[3] = (int)(((unsigned)NUM_CODES >> 16) | (128u << 16)); // dim1.hi16 | tile_dim0=128
    g1[4] = 64;                                            // tile_dim1=64, tile_dim2=0
    g1[5] = 128;                                           // tensor_dim0_stride = 128 B
    g1[6] = 0;
    g1[7] = 0;                                             // tensor_dim1_stride unused (2D)
    i32x4 z4 = {0, 0, 0, 0};
    i32x8 z8 = {0, 0, 0, 0, 0, 0, 0, 0};
    __builtin_amdgcn_tensor_load_to_lds(g0, g1, z4, z4, z8, 0);  // 6-arg form (clang-23)
}

// ---------------------------------------------------------------------------
// Kernel 1: per (64-token tile, code-chunk) partial argmin of (||c||^2 - 2 x.c)
// 4 waves/block; wave w owns token rows 16w..16w+15. TDM double-buffers tiles.
// ---------------------------------------------------------------------------
__global__ __launch_bounds__(128)
void nn_partial_kernel(const float* __restrict__ x,
                       const unsigned char* __restrict__ chi,
                       const unsigned char* __restrict__ clo,
                       const float* __restrict__ c2,
                       float* __restrict__ pmin, int* __restrict__ pidx) {
    __shared__ __align__(128) unsigned short xs_hi[BM][DIM];
    __shared__ __align__(128) unsigned short xs_lo[BM][DIM];
    __shared__ __align__(128) unsigned short cs_hi[2][BN][DIM];   // 2 x 8KB
    __shared__ __align__(128) unsigned short cs_lo[2][BN][DIM];   // 2 x 8KB

    const int tid   = threadIdx.x;     // 0..127
    const int wave  = tid >> 5;        // 0..3
    const int lane  = tid & 31;
    const int mTile = blockIdx.x;      // 0..127
    const int chunk = blockIdx.y;      // 0..7

    const int chunkBeg = chunk * (NUM_CODES / NSPLIT);
    const int nIter    = (NUM_CODES / NSPLIT + BN - 1) / BN;     // 98 tiles of 64 codes

    // ---- stage x tile (64 tokens x 64 dims) as bf16 hi/lo (one-time) ----
    {
        const int row  = tid >> 1;
        const int col0 = (tid & 1) * 32;
        const float* src = x + (size_t)(mTile * BM + row) * DIM + col0;
        #pragma unroll 8
        for (int j = 0; j < 32; ++j) {
            float f = src[j];
            unsigned short h = bf16_rne(f);
            float hf = __uint_as_float((unsigned)h << 16);
            xs_hi[row][col0 + j] = h;
            xs_lo[row][col0 + j] = bf16_rne(f - hf);
        }
    }
    __syncthreads();

    // ---- per-wave A fragments (16x32 bf16; kstep s covers K = 32s..32s+31)
    // lanes 0-15 : M=lane,    K in {0..15}+32s ; lanes16-31: M=lane-16, K in {16..31}+32s
    const int mLocal = (wave << 4) + (lane & 15);
    const int kGrp   = (lane >> 4) * 16;
    AFrag a_hi[2], a_lo[2];
    #pragma unroll
    for (int s = 0; s < 2; ++s) {
        const int kb = s * 32 + kGrp;
        a_hi[s].h[0] = *reinterpret_cast<const v8bf*>(&xs_hi[mLocal][kb]);
        a_hi[s].h[1] = *reinterpret_cast<const v8bf*>(&xs_hi[mLocal][kb + 8]);
        a_lo[s].h[0] = *reinterpret_cast<const v8bf*>(&xs_lo[mLocal][kb]);
        a_lo[s].h[1] = *reinterpret_cast<const v8bf*>(&xs_lo[mLocal][kb + 8]);
    }

    float minv[8];
    int   mini[8];
    #pragma unroll
    for (int i = 0; i < 8; ++i) { minv[i] = 3.4e38f; mini[i] = 0; }

    const int nInLane = lane & 15;          // N within a 16-wide tile (C/D layout)
    const int bkGrp   = (lane >> 4) * 16;   // B: lane group selects K half

    const unsigned ldsHi[2] = { (unsigned)(size_t)&cs_hi[0][0][0], (unsigned)(size_t)&cs_hi[1][0][0] };
    const unsigned ldsLo[2] = { (unsigned)(size_t)&cs_lo[0][0][0], (unsigned)(size_t)&cs_lo[1][0][0] };
    const unsigned long long gHi = (unsigned long long)(size_t)chi + (unsigned long long)chunkBeg * 128ull;
    const unsigned long long gLo = (unsigned long long)(size_t)clo + (unsigned long long)chunkBeg * 128ull;

    // prologue: stream tile 0 into buffer 0 (wave 0 drives the TDM)
    if (wave == 0) {
        tdm_load_tile(gHi, ldsHi[0]);
        tdm_load_tile(gLo, ldsLo[0]);
    }

    for (int it = 0; it < nIter; ++it) {
        const int cur = it & 1;
        __syncthreads();                         // buf[cur^1] free (prev compute done)
        if (wave == 0) {
            if (it + 1 < nIter) {
                const unsigned long long off = (unsigned long long)(it + 1) * (BN * 128ull);
                tdm_load_tile(gHi + off, ldsHi[cur ^ 1]);
                tdm_load_tile(gLo + off, ldsLo[cur ^ 1]);
                __builtin_amdgcn_s_wait_tensorcnt(2);   // tile `it` complete, next in flight
            } else {
                __builtin_amdgcn_s_wait_tensorcnt(0);
            }
        }
        __syncthreads();                         // buf[cur] visible to all waves

        const unsigned short (*chiT)[DIM] = cs_hi[cur];
        const unsigned short (*cloT)[DIM] = cs_lo[cur];
        const int tileBase = chunkBeg + it * BN;

        // prefetch the 4 tiles' ||c||^2 values up front (hides latency behind WMMAs,
        // fills the WMMA->VALU hazard slots with useful work instead of v_nops)
        float c2v[4];
        #pragma unroll
        for (int t = 0; t < 4; ++t)
            c2v[t] = c2[tileBase + t * 16 + nInLane];   // +INF-padded: no guard

        #pragma unroll
        for (int t = 0; t < 4; ++t) {
            const int nLocal = t * 16 + nInLane;
            v8f acc = {};
            #pragma unroll
            for (int s = 0; s < 2; ++s) {
                const v16bf b_hi = *reinterpret_cast<const v16bf*>(&chiT[nLocal][s * 32 + bkGrp]);
                const v16bf b_lo = *reinterpret_cast<const v16bf*>(&cloT[nLocal][s * 32 + bkGrp]);
                acc = __builtin_amdgcn_wmma_f32_16x16x32_bf16(false, a_lo[s].v, false, b_hi,
                                                              (short)0, acc, false, false);
                acc = __builtin_amdgcn_wmma_f32_16x16x32_bf16(false, a_hi[s].v, false, b_lo,
                                                              (short)0, acc, false, false);
                acc = __builtin_amdgcn_wmma_f32_16x16x32_bf16(false, a_hi[s].v, false, b_hi,
                                                              (short)0, acc, false, false);
            }
            const int gcode = tileBase + nLocal;
            #pragma unroll
            for (int i = 0; i < 8; ++i) {        // VGPR i -> M = i + 8*(lane>=16)
                float d = __builtin_fmaf(-2.0f, acc[i], c2v[t]);
                if (d < minv[i]) { minv[i] = d; mini[i] = gcode; }
            }
        }
    }

    // ---- reduce (min, idx) across the 16 lanes that share each M row ----
    #pragma unroll
    for (int i = 0; i < 8; ++i) {
        float v = minv[i]; int ix = mini[i];
        #pragma unroll
        for (int m = 1; m < 16; m <<= 1) {
            float vo = __shfl_xor(v, m, 32);
            int   io = __shfl_xor(ix, m, 32);
            if (vo < v || (vo == v && io < ix)) { v = vo; ix = io; }
        }
        minv[i] = v; mini[i] = ix;
    }
    if ((lane & 15) == 0) {
        const int rowBase = mTile * BM + (wave << 4) + (lane >> 4) * 8;
        #pragma unroll
        for (int i = 0; i < 8; ++i) {
            pmin[(size_t)chunk * NTOK + rowBase + i] = minv[i];
            pidx[(size_t)chunk * NTOK + rowBase + i] = mini[i];
        }
    }
}

// ---------------------------------------------------------------------------
// Kernel 2: fold chunk partials, add ||x||^2, apply threshold, emit int32 ids
// ---------------------------------------------------------------------------
__global__ __launch_bounds__(256)
void nn_finalize_kernel(const float* __restrict__ x, const float* __restrict__ pmin,
                        const int* __restrict__ pidx, int* __restrict__ out) {
    const int tok = blockIdx.x * blockDim.x + threadIdx.x;
    if (tok >= NTOK) return;
    const float* xr = x + (size_t)tok * DIM;
    float x2 = 0.0f;
    #pragma unroll 8
    for (int k = 0; k < DIM; ++k) { float f = xr[k]; x2 = __builtin_fmaf(f, f, x2); }
    float best = 3.4e38f; int bi = 0;
    #pragma unroll
    for (int c = 0; c < NSPLIT; ++c) {           // ascending chunks: first-min semantics
        float v = pmin[(size_t)c * NTOK + tok];
        int  ix = pidx[(size_t)c * NTOK + tok];
        if (v < best) { best = v; bi = ix; }
    }
    out[tok] = ((x2 + best) <= THRESH) ? bi : -1;
}

extern "C" void kernel_launch(void* const* d_in, const int* in_sizes, int n_in,
                              void* d_out, int out_size, void* d_ws, size_t ws_size,
                              hipStream_t stream) {
    (void)in_sizes; (void)n_in; (void)out_size; (void)ws_size;
    const float* x     = (const float*)d_in[0];
    const float* codes = (const float*)d_in[1];

    // workspace layout (~13.6 MB): chi | clo | c2(+pad) | pmin | pidx
    char* ws = (char*)d_ws;
    unsigned* chi = (unsigned*)ws;                               ws += (size_t)NUM_CODES * DIM * 2;
    unsigned* clo = (unsigned*)ws;                               ws += (size_t)NUM_CODES * DIM * 2;
    float*    c2  = (float*)ws;                                  ws += (size_t)(NUM_CODES + BN) * 4;
    float*    pmin = (float*)ws;                                 ws += (size_t)NSPLIT * NTOK * 4;
    int*      pidx = (int*)ws;

    code_precompute_kernel<<<(NUM_CODES + BN + 127) / 128, 256, 0, stream>>>(codes, chi, clo, c2);
    dim3 grid1(NTOK / BM, NSPLIT);   // 128 x 8 = 1024 workgroups
    nn_partial_kernel<<<grid1, 128, 0, stream>>>(x, (const unsigned char*)chi,
                                                 (const unsigned char*)clo, c2, pmin, pidx);
    nn_finalize_kernel<<<(NTOK + 255) / 256, 256, 0, stream>>>(x, pmin, pidx, (int*)d_out);
}
